// VQModel_66563403153722
// MI455X (gfx1250) — matrix operand order repaired and checked by reading, not compiled, not run on hardware
//
#include <hip/hip_runtime.h>
#include <hip/hip_bf16.h>
#include <math.h>

typedef __attribute__((ext_vector_type(16))) _Float16 v16h;
typedef __attribute__((ext_vector_type(8)))  float    v8f;
typedef __attribute__((__vector_size__(4 * sizeof(int)))) int i32x4;

#define DIM    256
#define KC     16384
#define MTOK   8192
#define DECAYF 0.99f
#define OMD    0.01f
#define EPSF   1e-5f

// padded LDS row: 256 data halves + 8 pad halves -> 528B stride (4-bank skew, conflict-free)
#define BROW   264

typedef union { float4 q[2]; v16h v; } h16u;

#if defined(__gfx1250__) && __has_builtin(__builtin_amdgcn_global_load_async_to_lds_b128)
#define HAVE_ASYNC_LDS 1
#else
#define HAVE_ASYNC_LDS 0
#endif

__device__ __forceinline__ float waveReduceSum(float v) {
#pragma unroll
  for (int m = 16; m >= 1; m >>= 1) v += __shfl_xor(v, m, 32);
  return v;
}

// monotonic float -> uint key (total order matches float <)
__device__ __forceinline__ unsigned int fkey(float f) {
  unsigned int b = __float_as_uint(f);
  return (b & 0x80000000u) ? ~b : (b | 0x80000000u);
}

__device__ __forceinline__ void asyncCopy16(const _Float16* g, _Float16* l) {
#if HAVE_ASYNC_LDS
  __builtin_amdgcn_global_load_async_to_lds_b128(
      (__attribute__((address_space(1))) i32x4*)g,
      (__attribute__((address_space(3))) i32x4*)l, 0, 0);
#else
  *(float4*)l = *(const float4*)g;
#endif
}

__device__ __forceinline__ void waitAsync() {
#if HAVE_ASYNC_LDS
#if __has_builtin(__builtin_amdgcn_s_wait_asynccnt)
  __builtin_amdgcn_s_wait_asynccnt(0);
#else
  asm volatile("s_wait_asynccnt 0" ::: "memory");
#endif
#endif
}

// ---------------- normalize z: z_n f32 + f16, zn2 ----------------
__global__ __launch_bounds__(256) void norm_z_kernel(const float* __restrict__ z,
                                                     float* __restrict__ znF,
                                                     _Float16* __restrict__ znH,
                                                     float* __restrict__ zn2) {
  int lane = threadIdx.x & 31;
  int m = blockIdx.x * 8 + (threadIdx.x >> 5);
  const float4* zr = (const float4*)(z + (size_t)m * DIM);
  float4 x0 = zr[lane];
  float4 x1 = zr[lane + 32];
  float s = x0.x * x0.x + x0.y * x0.y + x0.z * x0.z + x0.w * x0.w +
            x1.x * x1.x + x1.y * x1.y + x1.z * x1.z + x1.w * x1.w;
  s = waveReduceSum(s);
  float inv = 1.0f / fmaxf(sqrtf(s), 1e-12f);
  float4 y0 = make_float4(x0.x * inv, x0.y * inv, x0.z * inv, x0.w * inv);
  float4 y1 = make_float4(x1.x * inv, x1.y * inv, x1.z * inv, x1.w * inv);
  ((float4*)(znF + (size_t)m * DIM))[lane] = y0;
  ((float4*)(znF + (size_t)m * DIM))[lane + 32] = y1;
  union { float2 f2; _Float16 h[4]; } p0, p1;
  p0.h[0] = (_Float16)y0.x; p0.h[1] = (_Float16)y0.y; p0.h[2] = (_Float16)y0.z; p0.h[3] = (_Float16)y0.w;
  p1.h[0] = (_Float16)y1.x; p1.h[1] = (_Float16)y1.y; p1.h[2] = (_Float16)y1.z; p1.h[3] = (_Float16)y1.w;
  ((float2*)(znH + (size_t)m * DIM))[lane] = p0.f2;
  ((float2*)(znH + (size_t)m * DIM))[lane + 32] = p1.f2;
  if (lane == 0) zn2[m] = s * inv * inv;
}

// ---------------- normalize codebook: e_n f16, en2 ----------------
__global__ __launch_bounds__(256) void norm_e_kernel(const float* __restrict__ cb,
                                                     _Float16* __restrict__ enH,
                                                     float* __restrict__ en2) {
  int lane = threadIdx.x & 31;
  int k = blockIdx.x * 8 + (threadIdx.x >> 5);
  const float4* er = (const float4*)(cb + (size_t)k * DIM);
  float4 x0 = er[lane];
  float4 x1 = er[lane + 32];
  float s = x0.x * x0.x + x0.y * x0.y + x0.z * x0.z + x0.w * x0.w +
            x1.x * x1.x + x1.y * x1.y + x1.z * x1.z + x1.w * x1.w;
  s = waveReduceSum(s);
  float inv = 1.0f / fmaxf(sqrtf(s), 1e-12f);
  union { float2 f2; _Float16 h[4]; } p0, p1;
  p0.h[0] = (_Float16)(x0.x * inv); p0.h[1] = (_Float16)(x0.y * inv);
  p0.h[2] = (_Float16)(x0.z * inv); p0.h[3] = (_Float16)(x0.w * inv);
  p1.h[0] = (_Float16)(x1.x * inv); p1.h[1] = (_Float16)(x1.y * inv);
  p1.h[2] = (_Float16)(x1.z * inv); p1.h[3] = (_Float16)(x1.w * inv);
  ((float2*)(enH + (size_t)k * DIM))[lane] = p0.f2;
  ((float2*)(enH + (size_t)k * DIM))[lane + 32] = p1.f2;
  if (lane == 0) en2[k] = s * inv * inv;
}

// ---------------- WMMA GEMM: d = zn2 + en2 - 2*z_n.e_n^T, fused argmin ----------------
// block = 128-row M supertile x 1024-code K slice; 8 waves share a double-buffered
// LDS B tile staged with GLOBAL_LOAD_ASYNC_TO_LDS_B128 (overlap DMA(ct+1) with WMMA(ct)).
__global__ __launch_bounds__(256) void gemm_argmin_kernel(const _Float16* __restrict__ znH,
                                                          const _Float16* __restrict__ enH,
                                                          const float* __restrict__ zn2,
                                                          const float* __restrict__ en2,
                                                          float* __restrict__ out_d,
                                                          unsigned long long* __restrict__ amin) {
  __shared__ _Float16 lds[2][16 * BROW];

  int lane = threadIdx.x & 31;
  int wid = threadIdx.x >> 5;
  int super = blockIdx.x >> 4;          // 64 supertiles of 128 rows
  int k0 = (blockIdx.x & 15) * 1024;    // 16 slices of 1024 codes
  int m0 = super * 128 + wid * 16;
  int hh = lane >> 4;
  int ln = lane & 15;

  // staging: chunk c (of 512 per tile): code = c>>5, 16B sub-chunk w = c&31
  int c0 = threadIdx.x;
  int code0 = c0 >> 5, w0 = c0 & 31;

  // A tile: 16 rows x 256 dims f16, resident in 64 VGPRs
  h16u a[8];
  const float4* ar = (const float4*)(znH + (size_t)(m0 + ln) * DIM);
#pragma unroll
  for (int s = 0; s < 8; ++s) {
    a[s].q[0] = ar[4 * s + 2 * hh];
    a[s].q[1] = ar[4 * s + 2 * hh + 1];
  }

  float zrow[8];
#pragma unroll
  for (int j = 0; j < 8; ++j) zrow[j] = zn2[m0 + j + 8 * hh];

  float rmin[8];
  int ridx[8];
#pragma unroll
  for (int j = 0; j < 8; ++j) { rmin[j] = 3.4e38f; ridx[j] = 0; }

  // prologue: stage tile 0 into buffer 0
  {
    int cb = k0;
    asyncCopy16(enH + (size_t)(cb + code0) * DIM + w0 * 8, &lds[0][code0 * BROW + w0 * 8]);
    asyncCopy16(enH + (size_t)(cb + code0 + 8) * DIM + w0 * 8, &lds[0][(code0 + 8) * BROW + w0 * 8]);
  }
  waitAsync();
  __syncthreads();

  for (int ct = 0; ct < 64; ++ct) {
    int cb = k0 + ct * 16;
    const _Float16* cur = lds[ct & 1];

    // kick off DMA for next tile into the other buffer (its last readers
    // finished at the barrier ending iteration ct-1)
    if (ct + 1 < 64) {
      int nb = k0 + (ct + 1) * 16;
      _Float16* nxt = lds[(ct + 1) & 1];
      asyncCopy16(enH + (size_t)(nb + code0) * DIM + w0 * 8, &nxt[code0 * BROW + w0 * 8]);
      asyncCopy16(enH + (size_t)(nb + code0 + 8) * DIM + w0 * 8, &nxt[(code0 + 8) * BROW + w0 * 8]);
    }

    v8f acc = {0.f, 0.f, 0.f, 0.f, 0.f, 0.f, 0.f, 0.f};
#pragma unroll
    for (int s = 0; s < 8; ++s) {
      h16u b;
      const float4* bp = (const float4*)(cur + ln * BROW + 32 * s + 16 * hh);
      b.q[0] = bp[0];
      b.q[1] = bp[1];
      acc = __builtin_amdgcn_wmma_f32_16x16x32_f16(false, a[s].v, false, b.v,
                                                   (short)0, acc, false, false);
    }

    float e2 = en2[cb + ln];
#pragma unroll
    for (int j = 0; j < 8; ++j) {
      float dj = zrow[j] + e2 - 2.0f * acc[j];
      __builtin_nontemporal_store(dj, out_d + (size_t)(m0 + j + 8 * hh) * KC + (cb + ln));
      if (dj < rmin[j]) { rmin[j] = dj; ridx[j] = cb + ln; }
    }

    waitAsync();        // next tile landed while we computed
    __syncthreads();
  }

  // half-wave (16-lane) min+argmin reduction, then packed u64 atomicMin per row
#pragma unroll
  for (int j = 0; j < 8; ++j) {
    float v = rmin[j];
    int id = ridx[j];
#pragma unroll
    for (int msk = 8; msk >= 1; msk >>= 1) {
      float vo = __shfl_xor(v, msk, 32);
      int io = __shfl_xor(id, msk, 32);
      if (vo < v || (vo == v && io < id)) { v = vo; id = io; }
    }
    if (ln == 0) {
      unsigned long long pk = ((unsigned long long)fkey(v) << 32) | (unsigned int)id;
      atomicMin(amin + (m0 + j + 8 * hh), pk);
    }
  }
}

// ---------------- decode argmin, counts ----------------
__global__ __launch_bounds__(256) void decode_kernel(const unsigned long long* __restrict__ amin,
                                                     int* __restrict__ idxInt,
                                                     float* __restrict__ out_idx,
                                                     float* __restrict__ counts) {
  int m = blockIdx.x * 256 + threadIdx.x;
  if (m < MTOK) {
    unsigned long long pk = amin[m];
    int id = (int)(unsigned int)(pk & 0xffffffffull);
    idxInt[m] = id;
    out_idx[m] = (float)id;
    atomicAdd(counts + id, 1.0f);
  }
}

// ---------------- z_q gather + loss partial ----------------
__global__ __launch_bounds__(256) void zq_loss_kernel(const float* __restrict__ codebook,
                                                      const float* __restrict__ z,
                                                      const int* __restrict__ idxInt,
                                                      float* __restrict__ out_zq,
                                                      float* __restrict__ scal) {
  int m = blockIdx.x;
  int t = threadIdx.x;
  int k = idxInt[m];
  size_t o = (size_t)m * DIM + t;
  float q = codebook[(size_t)k * DIM + t];
  out_zq[o] = q;
  float df = q - z[o];
  float s = waveReduceSum(df * df);
  __shared__ float red[8];
  if ((threadIdx.x & 31) == 0) red[threadIdx.x >> 5] = s;
  __syncthreads();
  if (threadIdx.x == 0) {
    float tt = 0.f;
#pragma unroll
    for (int i = 0; i < 8; ++i) tt += red[i];
    atomicAdd(&scal[0], tt);
  }
}

// ---------------- embed_sum = segment_sum(z_n, idx) ----------------
__global__ __launch_bounds__(256) void embed_sum_kernel(const int* __restrict__ idxInt,
                                                        const float* __restrict__ znF,
                                                        float* __restrict__ embSum) {
  int m = blockIdx.x;
  int t = threadIdx.x;
  int k = idxInt[m];
  atomicAdd(embSum + (size_t)k * DIM + t, znF[(size_t)m * DIM + t]);
}

// ---------------- cluster_size EMA + n / perplexity partials ----------------
__global__ __launch_bounds__(256) void cluster_kernel(const float* __restrict__ counts,
                                                      const float* __restrict__ cs_in,
                                                      float* __restrict__ out_cs,
                                                      float* __restrict__ scal) {
  int k = blockIdx.x * 256 + threadIdx.x;
  float cnt = counts[k];
  float csn = cs_in[k] * DECAYF + OMD * cnt;
  out_cs[k] = csn;
  float p = cnt * (1.0f / (float)MTOK);
  float pl = p * logf(p + 1e-7f);
  float s1 = waveReduceSum(csn);
  float s2 = waveReduceSum(pl);
  __shared__ float red[16];
  int w = threadIdx.x >> 5;
  if ((threadIdx.x & 31) == 0) { red[w] = s1; red[8 + w] = s2; }
  __syncthreads();
  if (threadIdx.x == 0) {
    float a = 0.f, b = 0.f;
#pragma unroll
    for (int i = 0; i < 8; ++i) { a += red[i]; b += red[8 + i]; }
    atomicAdd(&scal[1], a);
    atomicAdd(&scal[2], b);
  }
}

// ---------------- scalar finalize: loss, perplexity ----------------
__global__ void finalize_kernel(const float* __restrict__ scal,
                                float* __restrict__ out_loss,
                                float* __restrict__ out_perp) {
  if (threadIdx.x == 0) {
    out_loss[0] = scal[0] * (1.0f / (float)(MTOK * DIM));
    out_perp[0] = expf(-scal[2]);
  }
}

// ---------------- embed_avg EMA + new_weight ----------------
__global__ __launch_bounds__(256) void embed_final_kernel(const float* __restrict__ ea_in,
                                                          const float* __restrict__ embSum,
                                                          const float* __restrict__ out_cs,
                                                          const float* __restrict__ scal,
                                                          float* __restrict__ out_ea,
                                                          float* __restrict__ out_w) {
  size_t i = (size_t)blockIdx.x * 256 + threadIdx.x;
  float ea = ea_in[i] * DECAYF + OMD * embSum[i];
  out_ea[i] = ea;
  int k = (int)(i >> 8);
  float n = scal[1];
  float csn = out_cs[k];
  float sm = (csn + EPSF) / (n + (float)KC * EPSF) * n;
  out_w[i] = ea / sm;
}

extern "C" void kernel_launch(void* const* d_in, const int* in_sizes, int n_in,
                              void* d_out, int out_size, void* d_ws, size_t ws_size,
                              hipStream_t stream) {
  const float* z            = (const float*)d_in[0];  // [8192,256]
  const float* codebook     = (const float*)d_in[1];  // [16384,256]
  const float* cluster_size = (const float*)d_in[2];  // [16384]
  const float* embed_avg    = (const float*)d_in[3];  // [16384,256]

  float* out = (float*)d_out;
  float* o_zq   = out;                 // 2097152
  float* o_loss = out + 2097152;       // 1
  float* o_d    = out + 2097153;       // 134217728
  float* o_idx  = out + 136314881;     // 8192
  float* o_perp = out + 136323073;     // 1
  float* o_cs   = out + 136323074;     // 16384
  float* o_ea   = out + 136339458;     // 4194304
  float* o_w    = out + 140533762;     // 4194304

  // workspace layout (float offsets)
  float* ws = (float*)d_ws;
  float* counts = ws;                                    // 16384
  float* scal   = ws + 16384;                            // 8 scalars
  float* embSum = ws + 16392;                            // 4194304
  unsigned long long* amin = (unsigned long long*)(ws + 4210696); // 8192 u64
  int*   idxInt = (int*)(ws + 4227080);                  // 8192
  float* zn2    = ws + 4235272;                          // 8192
  float* en2    = ws + 4243464;                          // 16384
  float* znF    = ws + 4259848;                          // 2097152
  _Float16* znH = (_Float16*)(ws + 6357000);             // 2097152 halves
  _Float16* enH = (_Float16*)(ws + 7405576);             // 4194304 halves
  // total ws usage: 9502728 floats (~38 MB)

  (void)hipMemsetAsync(ws, 0, (size_t)4210696 * sizeof(float), stream);  // counts+scal+embSum
  (void)hipMemsetAsync(amin, 0xFF, (size_t)MTOK * sizeof(unsigned long long), stream);

  norm_z_kernel<<<1024, 256, 0, stream>>>(z, znF, znH, zn2);
  norm_e_kernel<<<2048, 256, 0, stream>>>(codebook, enH, en2);
  gemm_argmin_kernel<<<1024, 256, 0, stream>>>(znH, enH, zn2, en2, o_d, amin);
  decode_kernel<<<32, 256, 0, stream>>>(amin, idxInt, o_idx, counts);
  zq_loss_kernel<<<8192, 256, 0, stream>>>(codebook, z, idxInt, o_zq, scal);
  embed_sum_kernel<<<8192, 256, 0, stream>>>(idxInt, znF, embSum);
  cluster_kernel<<<64, 256, 0, stream>>>(counts, cluster_size, o_cs, scal);
  finalize_kernel<<<1, 32, 0, stream>>>(scal, o_loss, o_perp);
  embed_final_kernel<<<16384, 256, 0, stream>>>(embed_avg, embSum, o_cs, scal, o_ea, o_w);
}
